// Attention_I_46566035423949
// MI455X (gfx1250) — matrix operand order repaired and checked by reading, not compiled, not run on hardware
//
#include <hip/hip_runtime.h>

typedef __attribute__((ext_vector_type(16))) __bf16 v16bf;
typedef __attribute__((ext_vector_type(8)))  float  v8f;
typedef int v4i_t __attribute__((ext_vector_type(4)));

#define NTOK  15
#define SCALE 0.17677669529663687f  /* 1/sqrt(32) */

#if defined(__HIP_DEVICE_COMPILE__) && __has_builtin(__builtin_amdgcn_global_load_async_to_lds_b128)
#define USE_ASYNC_LDS 1
#else
#define USE_ASYNC_LDS 0
#endif

__device__ __forceinline__ __bf16 f2bf(float f) {
  union { float f; unsigned u; } v; v.f = f;
  unsigned r = v.u + 0x7FFFu + ((v.u >> 16) & 1u);   // round-to-nearest-even
  unsigned short h = (unsigned short)(r >> 16);
  return __builtin_bit_cast(__bf16, h);
}

// A fragment for 16-bit A 16x32, contiguous k order (k = 16*(lane/16) + s):
// lanes 0-15: M=lane, K = 0..15
// lanes 16-31: M=lane-16, K = 16..31
__device__ __forceinline__ v16bf load_frag_A(const __bf16* m, int stride, int lane) {
  const int M   = lane & 15;
  const int klo = (lane >> 4) * 16;
  const __bf16* p = m + M * stride;
  v16bf a;
#pragma unroll
  for (int t = 0; t < 8; ++t) a[t] = p[klo + t];
#pragma unroll
  for (int t = 0; t < 8; ++t) a[8 + t] = p[klo + 8 + t];
  return a;
}

// B fragment where source is stored [N][K] (one row per output column, K contiguous):
// lanes 0-15: col n=lane, K=0..15 ; lanes 16-31: col n=lane-16, K=16..31
__device__ __forceinline__ v16bf load_frag_B_nk(const __bf16* m, int stride, int lane) {
  const int n   = lane & 15;
  const int klo = (lane >> 4) * 16;
  const __bf16* p = m + n * stride + klo;
  v16bf b;
#pragma unroll
  for (int t = 0; t < 16; ++t) b[t] = p[t];
  return b;
}

// ---------------- prep kernels ----------------
// W [512][Nout] f32 (row-major, [in,out]) -> Wt [Nout][512] bf16
__global__ void transpose_w_bf16(const float* __restrict__ W, unsigned short* __restrict__ Wtu, int Nout) {
  int i = blockIdx.x * blockDim.x + threadIdx.x;
  if (i < 512 * Nout) {
    int k = i / Nout, n = i % Nout;
    ((__bf16*)Wtu)[(size_t)n * 512 + k] = f2bf(W[i]);
  }
}

// W_iconv [32][15] f32 -> WcT [16][32] bf16 (transposed, fragment-friendly [N][K],
// row 15 zeroed = padded key slot). Lane B-fragments become contiguous 32B loads.
__global__ void prep_wconv_bf16(const float* __restrict__ W, unsigned short* __restrict__ Wcu) {
  int i = threadIdx.x;
  if (i < 512) {
    int n = i >> 5, k = i & 31;
    ((__bf16*)Wcu)[i] = (n < NTOK) ? f2bf(W[k * NTOK + n]) : f2bf(0.0f);
  }
}

// ---------------- GEMM v2 ----------------
// Block = 128 threads (4 waves), block tile 64M x 64N, wave tile 64M x 16N
// (4 accumulators per wave reusing one B fragment per k-step).
// A source: f32 (fused bf16 conversion into LDS stage) or bf16 (async-to-LDS), by template.
template <bool A_IS_F32>
__global__ __launch_bounds__(128) void gemm_wmma_v2(
    const void* __restrict__ Aptr, const unsigned short* __restrict__ Wtu,
    float* __restrict__ outF, unsigned short* __restrict__ outBu,
    const float* __restrict__ bias, int Nout)
{
  constexpr int K  = 512;
  constexpr int AS = 520;                        // padded row stride (bank spread)
  __shared__ __align__(16) __bf16 As[64 * AS];   // ~65 KB of the 320 KB WGP LDS
  const __bf16* Wt = (const __bf16*)Wtu;
  __bf16* outB = (__bf16*)outBu;

  const int tid  = threadIdx.x;
  const int lane = tid & 31;
  const int wave = tid >> 5;
  const int tileM = blockIdx.x;   // M/64 tiles
  const int tileN = blockIdx.y;   // Nout/64 tiles (16 per wave)

  // Stage 64-row, full-K A tile into LDS
  if constexpr (A_IS_F32) {
    // f32 source: load + inline f32->bf16 convert + ds_store
    const float4* src = (const float4*)((const float*)Aptr + (size_t)tileM * 64 * K);
    for (int i = tid; i < 64 * 128; i += 128) {          // 4 floats per chunk
      const int row = i >> 7, c = i & 127;
      float4 v4 = src[(size_t)row * 128 + c];
      __bf16* d = As + row * AS + c * 4;
      d[0] = f2bf(v4.x); d[1] = f2bf(v4.y); d[2] = f2bf(v4.z); d[3] = f2bf(v4.w);
    }
  } else {
    const __bf16* base = (const __bf16*)Aptr + (size_t)tileM * 64 * K;
#if USE_ASYNC_LDS
    // bf16 source: asynchronous copy global->LDS (ASYNCcnt-tracked, no VGPR landing)
    typedef __attribute__((address_space(1))) v4i_t* gvp_t;
    typedef __attribute__((address_space(3))) v4i_t* lvp_t;
    for (int i = tid; i < 64 * 64; i += 128) {           // 16B per lane per op
      const int row = i >> 6, c = i & 63;
      gvp_t g = (gvp_t)(__bf16*)(base + (size_t)row * K + c * 8);
      lvp_t l = (lvp_t)(unsigned)(unsigned long long)(void*)(As + row * AS + c * 8);
      __builtin_amdgcn_global_load_async_to_lds_b128(g, l, 0, 0);
    }
    asm volatile("s_wait_asynccnt 0" ::: "memory");
#else
    const uint4* src = (const uint4*)base;
    for (int i = tid; i < 64 * 64; i += 128) {           // 8 bf16 per chunk
      const int row = i >> 6, c = i & 63;
      *((uint4*)(As + row * AS) + c) = src[(size_t)row * 64 + c];
    }
#endif
  }
  __syncthreads();

  const int nCol = tileN * 64 + wave * 16 + (lane & 15);
  const __bf16* wrow = Wt + (size_t)nCol * K + (lane >> 4) * 16;
  const int aoff = (lane & 15) * AS + (lane >> 4) * 16;

  v8f acc[4] = {};
#pragma unroll
  for (int ks = 0; ks < 16; ++ks) {
    v16bf b;
    const __bf16* bp = wrow + ks * 32;
    if (ks < 15) __builtin_prefetch(bp + 32, 0, 3);
#pragma unroll
    for (int t = 0; t < 16; ++t) b[t] = bp[t];
#pragma unroll
    for (int s = 0; s < 4; ++s) {
      const __bf16* ap = As + s * 16 * AS + aoff + ks * 32;
      v16bf a;
#pragma unroll
      for (int t = 0; t < 8; ++t) a[t] = ap[t];
#pragma unroll
      for (int t = 0; t < 8; ++t) a[8 + t] = ap[8 + t];
      acc[s] = __builtin_amdgcn_wmma_f32_16x16x32_bf16(false, a, false, b,
                                                       (short)0, acc[s], false, false);
    }
  }

  // C/D layout: lane<16 -> rows r, col=lane ; lane>=16 -> rows 8+r, col=lane-16
#pragma unroll
  for (int s = 0; s < 4; ++s) {
    const int mbase = tileM * 64 + s * 16 + (lane >> 4) * 8;
    if (outF) {
      const float bv = bias ? bias[nCol] : 0.0f;
#pragma unroll
      for (int r = 0; r < 8; ++r)
        outF[(size_t)(mbase + r) * Nout + nCol] = acc[s][r] + bv;
    } else {
#pragma unroll
      for (int r = 0; r < 8; ++r)
        outB[(size_t)(mbase + r) * Nout + nCol] = f2bf(acc[s][r]);
    }
  }
}

// ---------------- fused per-(b,h) attention: scores + softmax + attn@V ----------------
__global__ __launch_bounds__(32) void attn_wmma(
    const unsigned short* __restrict__ Jqkvu, const unsigned short* __restrict__ Iqkvu,
    const unsigned short* __restrict__ Wcu, unsigned short* __restrict__ Xau)
{
  constexpr int MS = 40;                           // padded stride
  __shared__ __align__(16) __bf16 mats[6][16][MS]; // Jq,Jk,Jv,Iq,Ik,Iv (row 15 zeroed)
  __shared__ __align__(16) float  sc[16][16];
  __shared__ __align__(16) __bf16 attnP[16][MS];

  const __bf16* Jqkv = (const __bf16*)Jqkvu;
  const __bf16* Iqkv = (const __bf16*)Iqkvu;
  const __bf16* WcT  = (const __bf16*)Wcu;         // [16][32], fragment-friendly
  __bf16* Xa = (__bf16*)Xau;

  const int lane = threadIdx.x;
  const int b = blockIdx.x >> 4;
  const int h = blockIdx.x & 15;

  // 6 matrices x 16 rows x 4 sixteen-byte chunks, zero-fill padded row 15
  for (int c = lane; c < 6 * 16 * 4; c += 32) {
    const int m = c >> 6, row = (c >> 2) & 15, q = c & 3;
    uint4 val = make_uint4(0, 0, 0, 0);
    if (row < NTOK) {
      const __bf16* src = (m < 3 ? Jqkv : Iqkv)
          + (size_t)(b * NTOK + row) * 1536 + (m % 3) * 512 + h * 32 + q * 8;
      val = *(const uint4*)src;
    }
    *(uint4*)&mats[m][row][q * 8] = val;
  }
  __syncthreads();

  v16bf a_q  = load_frag_A(&mats[0][0][0], MS, lane);
  v16bf b_k  = load_frag_B_nk(&mats[1][0][0], MS, lane);  // k stored [m][d] == B^T rows
  v16bf a_iq = load_frag_A(&mats[3][0][0], MS, lane);
  v16bf b_ik = load_frag_B_nk(&mats[4][0][0], MS, lane);
  v16bf a_iv = load_frag_A(&mats[5][0][0], MS, lane);
  v16bf b_wc = load_frag_B_nk(WcT, 32, lane);             // 2 contiguous b128 loads

  v8f s = {};
  s = __builtin_amdgcn_wmma_f32_16x16x32_bf16(false, a_q,  false, b_k,  (short)0, s, false, false);
  s = __builtin_amdgcn_wmma_f32_16x16x32_bf16(false, a_iq, false, b_ik, (short)0, s, false, false);
  s = __builtin_amdgcn_wmma_f32_16x16x32_bf16(false, a_iv, false, b_wc, (short)0, s, false, false);

  const int colN  = lane & 15;
  const int rbase = (lane >> 4) * 8;
#pragma unroll
  for (int r = 0; r < 8; ++r) sc[rbase + r][colN] = s[r] * SCALE;
  __syncthreads();

  if (lane < 16) {
    if (lane < NTOK) {
      float mx = -3.0e38f;
      for (int m = 0; m < NTOK; ++m) mx = fmaxf(mx, sc[lane][m]);
      float e[NTOK], sum = 0.0f;
      for (int m = 0; m < NTOK; ++m) { e[m] = __expf(sc[lane][m] - mx); sum += e[m]; }
      const float inv = 1.0f / sum;
      for (int m = 0; m < NTOK; ++m) attnP[lane][m] = f2bf(e[m] * inv);
      for (int m = NTOK; m < 32; ++m) attnP[lane][m] = f2bf(0.0f);
    } else {
      for (int m = 0; m < 32; ++m) attnP[15][m] = f2bf(0.0f);
    }
  }
  __syncthreads();

  // x = attn(16x32, K=m padded) @ Jv(m x 32), d split into two 16-wide halves
  v16bf a_p = load_frag_A(&attnP[0][0], MS, lane);
#pragma unroll
  for (int half = 0; half < 2; ++half) {
    v16bf b_v;
    {
      const int d = half * 16 + (lane & 15);
      const int klo = (lane >> 4) * 16;
#pragma unroll
      for (int t = 0; t < 16; ++t) {
        const int m = klo + t;
        b_v[t] = (m < 16) ? mats[2][m][d] : f2bf(0.0f); // m>=16: A cols are zero anyway
      }
    }
    v8f x = {};
    x = __builtin_amdgcn_wmma_f32_16x16x32_bf16(false, a_p, false, b_v, (short)0, x, false, false);
#pragma unroll
    for (int r = 0; r < 8; ++r) {
      const int n = rbase + r;
      if (n < NTOK)
        Xa[(size_t)(b * NTOK + n) * 512 + h * 32 + half * 16 + colN] = f2bf(x[r]);
    }
  }
}

// ---------------- host launcher ----------------
extern "C" void kernel_launch(void* const* d_in, const int* in_sizes, int n_in,
                              void* d_out, int out_size, void* d_ws, size_t ws_size,
                              hipStream_t stream) {
  const float* jf    = (const float*)d_in[0];
  const float* itf   = (const float*)d_in[1];
  const float* Wjqkv = (const float*)d_in[2];
  const float* Wiqk  = (const float*)d_in[3];
  const float* Wic   = (const float*)d_in[4];
  const float* Wpr   = (const float*)d_in[5];
  const float* bpr   = (const float*)d_in[6];

  const size_t M = 4096 * 15; // 61440 token rows
  char* ws = (char*)d_ws;
  size_t o = 0;
  auto alloc = [&](size_t bytes) {
    char* p = ws + o;
    o += (bytes + 255) & ~(size_t)255;
    return p;
  };
  unsigned short* Jqkv = (unsigned short*)alloc(M * 1536 * 2);
  unsigned short* Iqkv = (unsigned short*)alloc(M * 1536 * 2);
  unsigned short* Xa   = (unsigned short*)alloc(M * 512 * 2);
  unsigned short* Wjt  = (unsigned short*)alloc(1536 * 512 * 2);
  unsigned short* Wit  = (unsigned short*)alloc(1536 * 512 * 2);
  unsigned short* Wpt  = (unsigned short*)alloc(512 * 512 * 2);
  unsigned short* Wcp  = (unsigned short*)alloc(16 * 32 * 2);

  transpose_w_bf16<<<(512 * 1536 + 255) / 256, 256, 0, stream>>>(Wjqkv, Wjt, 1536);
  transpose_w_bf16<<<(512 * 1536 + 255) / 256, 256, 0, stream>>>(Wiqk,  Wit, 1536);
  transpose_w_bf16<<<(512 * 512 + 255) / 256, 256, 0, stream>>>(Wpr, Wpt, 512);
  prep_wconv_bf16<<<1, 512, 0, stream>>>(Wic, Wcp);

  // QKV projections: f32 features in, bf16 qkv out (conversion fused into LDS stage)
  dim3 g1((unsigned)(M / 64), 1536 / 64);
  gemm_wmma_v2<true><<<g1, 128, 0, stream>>>(jf,  Wjt, nullptr, Jqkv, nullptr, 1536);
  gemm_wmma_v2<true><<<g1, 128, 0, stream>>>(itf, Wit, nullptr, Iqkv, nullptr, 1536);

  attn_wmma<<<4096 * 16, 32, 0, stream>>>(Jqkv, Iqkv, Wcp, Xa);

  // Output projection: bf16 A (async-to-LDS staging), f32 out + bias
  dim3 g2((unsigned)(M / 64), 512 / 64);
  gemm_wmma_v2<false><<<g2, 128, 0, stream>>>(Xa, Wpt, (float*)d_out, nullptr, bpr, 512);
}
